// FastFastFFlayer_80805514707485
// MI455X (gfx1250) — compile-verified, matrix operand order checked by reading, not run on hardware
//
#include <hip/hip_runtime.h>

// ---------------- CDNA5 WMMA types ----------------
typedef __attribute__((ext_vector_type(16))) __bf16 v16bf;
typedef __attribute__((ext_vector_type(8)))  float  v8f;

// Problem constants
#define SS   1024
#define BB   4
#define DD   512
#define HH   8
#define DHH  64
#define DVV  193            // 3*64+1
#define NPROJ 2576          // H*(5*DH+2) = 8*322
#define NPAD  2624          // 41*64, padded N for the projection GEMM
#define CHUNK 322           // per-head slice of qkvb row

__device__ __forceinline__ float elup1(float x) { return x > 0.f ? x + 1.f : __expf(x); }
__device__ __forceinline__ float sigm(float x)  { return 1.f / (1.f + __expf(-x)); }

__device__ __forceinline__ float wave_sum(float v) {
#pragma unroll
  for (int o = 16; o > 0; o >>= 1) v += __shfl_xor(v, o, 32);
  return v;
}

// ---- CDNA5 async copy: global -> LDS, tracked with ASYNCcnt ----
// One VMEM instruction per wave; each active lane moves one 16B chunk.
__device__ __forceinline__ void async_b128(unsigned lds_byte, const float* g) {
  asm volatile("global_load_async_to_lds_b128 %0, %1, off"
               :: "v"(lds_byte), "v"(g) : "memory");
}
__device__ __forceinline__ void wait_async_le1() {
  asm volatile("s_wait_asynccnt 0x1" ::: "memory");
}
__device__ __forceinline__ void wait_async_0() {
  asm volatile("s_wait_asynccnt 0x0" ::: "memory");
}

// Load a 16x32 bf16 WMMA fragment (A, or B^T with identical striping).
__device__ __forceinline__ v16bf load_frag(const __bf16* __restrict__ p) {
  v16bf f;
#pragma unroll
  for (int i = 0; i < 8; ++i) { f[i] = p[i]; f[i + 8] = p[i + 16]; }
  return f;
}

// ---------------- LayerNorm -> bf16 ----------------
__global__ __launch_bounds__(256) void ln_kernel(const float* __restrict__ x,
                                                 const float* __restrict__ g,
                                                 const float* __restrict__ b,
                                                 __bf16* __restrict__ xln) {
  const int lane = threadIdx.x & 31;
  const int row  = (blockIdx.x * blockDim.x + threadIdx.x) >> 5;  // 0..4095
  const float* xr = x + (size_t)row * DD;
  float s = 0.f, s2 = 0.f;
#pragma unroll
  for (int c = lane; c < DD; c += 32) { float v = xr[c]; s += v; s2 += v * v; }
  s = wave_sum(s); s2 = wave_sum(s2);
  const float m   = s * (1.f / DD);
  const float var = s2 * (1.f / DD) - m * m;
  const float r   = rsqrtf(var + 1e-5f);
#pragma unroll
  for (int c = lane; c < DD; c += 32)
    xln[(size_t)row * DD + c] = (__bf16)((xr[c] - m) * r * g[c] + b[c]);
}

// ---------------- weight transposes (f32 -> bf16, [N][K]) ----------------
__global__ __launch_bounds__(256) void tp_wslow_kernel(const float* __restrict__ w,
                                                       __bf16* __restrict__ wt) {
  const int idx = blockIdx.x * blockDim.x + threadIdx.x;  // NPAD*DD
  const int n = idx / DD, k = idx % DD;
  wt[(size_t)n * DD + k] = (n < NPROJ) ? (__bf16)w[(size_t)k * NPROJ + n] : (__bf16)0.f;
}

__global__ __launch_bounds__(256) void tp_wout_kernel(const float* __restrict__ w,
                                                      __bf16* __restrict__ wt) {
  const int idx = blockIdx.x * blockDim.x + threadIdx.x;  // DD*DD
  const int n = idx / DD, k = idx % DD;
  wt[(size_t)n * DD + k] = (__bf16)w[(size_t)k * DD + n];
}

// ---------------- projection GEMM: qkvb = xln @ w_slow (WMMA bf16) ----------------
__global__ __launch_bounds__(256) void gemm_qkvb_kernel(const __bf16* __restrict__ A,
                                                        const __bf16* __restrict__ Bt,
                                                        float* __restrict__ C) {
  const int lane = threadIdx.x & 31;
  const int task = blockIdx.x * 8 + (threadIdx.x >> 5);  // 256*41 tasks
  const int tn = task % 41, tm = task / 41;
  const int m0 = tm * 16, n0 = tn * 64;
  const int ksel = (lane >> 4) << 3;
  const int mrow = m0 + (lane & 15);
  v8f acc[4] = {v8f{}, v8f{}, v8f{}, v8f{}};
  for (int kk = 0; kk < DD; kk += 32) {
    const __bf16* ap = A + (size_t)mrow * DD + kk + ksel;
    __builtin_prefetch(ap + 32, 0, 1);
    v16bf af = load_frag(ap);
#pragma unroll
    for (int t = 0; t < 4; ++t) {
      const int n = n0 + t * 16 + (lane & 15);
      v16bf bf = load_frag(Bt + (size_t)n * DD + kk + ksel);
      acc[t] = __builtin_amdgcn_wmma_f32_16x16x32_bf16(false, af, false, bf,
                                                       (short)0, acc[t], false, false);
    }
  }
  const int nlo = lane & 15;
  const int mb  = m0 + ((lane >> 4) << 3);
#pragma unroll
  for (int t = 0; t < 4; ++t) {
    const int n = n0 + t * 16 + nlo;
    if (n < NPROJ)
#pragma unroll
      for (int r = 0; r < 8; ++r) C[(size_t)(mb + r) * NPROJ + n] = acc[t][r];
  }
}

// ---------------- act1: elu_p1_norm(q), elu_p1_norm(k), sigmoid(beta), in place ----
__global__ __launch_bounds__(256) void act1_kernel(float* __restrict__ qkvb) {
  const int lane = threadIdx.x & 31;
  const int task = (blockIdx.x * blockDim.x + threadIdx.x) >> 5;  // 4096*8
  const int h = task & 7, row = task >> 3;
  float* base = qkvb + (size_t)row * NPROJ + h * CHUNK;
#pragma unroll
  for (int part = 0; part < 2; ++part) {       // q then k
    float* p = base + part * DHH;
    float a0 = elup1(p[lane]), a1 = elup1(p[lane + 32]);
    float s = wave_sum(a0 + a1) + 1e-5f;
    p[lane] = a0 / s; p[lane + 32] = a1 / s;
  }
  if (lane == 0) base[321] = sigm(base[321]);
}

// ---------------- scan 1: delta rule, 64x193 state in LDS, async double-buffer ----
__global__ __launch_bounds__(256) void scan1_kernel(const float* __restrict__ qkvb,
                                                    const float* __restrict__ fw0,
                                                    float* __restrict__ out1,
                                                    float* __restrict__ fw_out) {
  __shared__ float Wsh[DHH * DVV];
  __shared__ __align__(16) float inbuf[2][328];   // 81 x 16B chunks + shift room
  const int tid = threadIdx.x;
  const int wv = tid >> 5, ln = tid & 31;
  const int bh = blockIdx.x;             // 0..31
  const int bb = bh >> 3, h = bh & 7;
  const int shift = (h & 1) ? 2 : 0;     // align DMA source down to 16B
  const unsigned ldsb = (unsigned)(unsigned long long)&inbuf[0][0];

  for (int i = tid; i < DHH * DVV; i += 256) Wsh[i] = fw0[(size_t)bh * DHH * DVV + i];

  const bool issuer = (wv < 3) && (ln < 27);      // 3 waves x 27 lanes = 81 chunks
  const int chunk = wv * 27 + ln;
  const float* row0 = qkvb + (size_t)bb * NPROJ + h * CHUNK - shift;
  if (issuer) async_b128(ldsb + 16u * chunk, row0 + 4 * chunk);

  for (int t = 0; t < SS; ++t) {
    __syncthreads();                     // prior compute done: buf[(t+1)&1] free
    if (t + 1 < SS) {
      if (issuer)
        async_b128(ldsb + ((unsigned)((t + 1) & 1)) * (328u * 4) + 16u * chunk,
                   qkvb + (size_t)((t + 1) * BB + bb) * NPROJ + h * CHUNK - shift + 4 * chunk);
      wait_async_le1();                  // step-t chunks have landed
    } else {
      wait_async_0();
    }
    __syncthreads();                     // publish buf[t&1] to all waves
    const float* in = &inbuf[t & 1][shift];
    if (tid < DVV) {
      const float sbeta = in[321];
      const float vj    = in[2 * DHH + tid];
      float vold = 0.f;
#pragma unroll 8
      for (int i = 0; i < DHH; ++i) vold += in[DHH + i] * Wsh[i * DVV + tid];
      const float coef = sbeta * (vj - vold);
      float acc = 0.f;
#pragma unroll 8
      for (int i = 0; i < DHH; ++i) {
        const float wvv = Wsh[i * DVV + tid] + in[DHH + i] * coef;  // rank-1 update
        Wsh[i * DVV + tid] = wvv;
        acc += in[i] * wvv;                                          // fused q^T W
      }
      out1[((size_t)bh * SS + t) * DVV + tid] = acc;
    }
  }
  __syncthreads();
  for (int i = tid; i < DHH * DVV; i += 256) fw_out[(size_t)bh * DHH * DVV + i] = Wsh[i];
}

// ---------------- act2: elu_p1_norm(fq), elu_p1_norm(fk), sigmoid(fb), in place ----
__global__ __launch_bounds__(256) void act2_kernel(float* __restrict__ out1) {
  const int lane = threadIdx.x & 31;
  const int task = (blockIdx.x * blockDim.x + threadIdx.x) >> 5;  // 32*1024
  float* base = out1 + (size_t)task * DVV;
#pragma unroll
  for (int part = 0; part < 2; ++part) {
    float* p = base + part * DHH;
    float a0 = elup1(p[lane]), a1 = elup1(p[lane + 32]);
    float s = wave_sum(a0 + a1) + 1e-5f;
    p[lane] = a0 / s; p[lane + 32] = a1 / s;
  }
  if (lane == 0) base[192] = sigm(base[192]);
}

// ---------------- scan 2: delta rule, 64x64 state in LDS, async double-buffer -----
__global__ __launch_bounds__(256) void scan2_kernel(const float* __restrict__ out1,
                                                    const float* __restrict__ vfw0,
                                                    __bf16* __restrict__ out2m,
                                                    float* __restrict__ vfw_out) {
  __shared__ float Wsh[DHH * DHH];
  __shared__ __align__(16) float inbuf[2][204];   // 50 x 16B chunks + shift room
  const int tid = threadIdx.x;
  const int wv = tid >> 5, ln = tid & 31;
  const int bh = blockIdx.x;
  const int bb = bh >> 3, h = bh & 7;
  const unsigned ldsb = (unsigned)(unsigned long long)&inbuf[0][0];

  for (int i = tid; i < DHH * DHH; i += 256) Wsh[i] = vfw0[(size_t)bh * DHH * DHH + i];

  const bool issuer = (wv < 2) && (ln < 25);      // 2 waves x 25 lanes = 50 chunks
  const int chunk = wv * 25 + ln;
  // row(t) starts at (bh*S+t)*193 floats; (193 % 4 == 1) -> shift(t) = t & 3
  if (issuer) async_b128(ldsb + 16u * chunk,
                         out1 + (size_t)bh * SS * DVV + 4 * chunk);  // t=0, shift=0

  for (int t = 0; t < SS; ++t) {
    __syncthreads();
    if (t + 1 < SS) {
      if (issuer) {
        const int shn = (t + 1) & 3;
        async_b128(ldsb + ((unsigned)((t + 1) & 1)) * (204u * 4) + 16u * chunk,
                   out1 + (size_t)(bh * SS + (t + 1)) * DVV - shn + 4 * chunk);
      }
      wait_async_le1();
    } else {
      wait_async_0();
    }
    __syncthreads();
    const float* in = &inbuf[t & 1][t & 3];
    if (tid < DHH) {
      const float sbeta = in[192];
      const float vj    = in[2 * DHH + tid];
      float vold = 0.f;
#pragma unroll 8
      for (int i = 0; i < DHH; ++i) vold += in[DHH + i] * Wsh[i * DHH + tid];
      const float coef = sbeta * (vj - vold);
      float acc = 0.f;
#pragma unroll 8
      for (int i = 0; i < DHH; ++i) {
        const float wvv = Wsh[i * DHH + tid] + in[DHH + i] * coef;
        Wsh[i * DHH + tid] = wvv;
        acc += in[i] * wvv;
      }
      out2m[(size_t)(t * BB + bb) * DD + h * DHH + tid] = (__bf16)acc;
    }
  }
  __syncthreads();
  for (int i = tid; i < DHH * DHH; i += 256) vfw_out[(size_t)bh * DHH * DHH + i] = Wsh[i];
}

// ---------------- output GEMM: y = x + out2 @ w_out (WMMA bf16) ----------------
__global__ __launch_bounds__(256) void gemm_out_kernel(const __bf16* __restrict__ A,
                                                       const __bf16* __restrict__ Bt,
                                                       const float* __restrict__ x,
                                                       float* __restrict__ y) {
  const int lane = threadIdx.x & 31;
  const int task = blockIdx.x * 8 + (threadIdx.x >> 5);  // 256*8 tasks
  const int tn = task & 7, tm = task >> 3;
  const int m0 = tm * 16, n0 = tn * 64;
  const int ksel = (lane >> 4) << 3;
  const int mrow = m0 + (lane & 15);
  v8f acc[4] = {v8f{}, v8f{}, v8f{}, v8f{}};
  for (int kk = 0; kk < DD; kk += 32) {
    v16bf af = load_frag(A + (size_t)mrow * DD + kk + ksel);
#pragma unroll
    for (int t = 0; t < 4; ++t) {
      const int n = n0 + t * 16 + (lane & 15);
      v16bf bf = load_frag(Bt + (size_t)n * DD + kk + ksel);
      acc[t] = __builtin_amdgcn_wmma_f32_16x16x32_bf16(false, af, false, bf,
                                                       (short)0, acc[t], false, false);
    }
  }
  const int nlo = lane & 15;
  const int mb  = m0 + ((lane >> 4) << 3);
#pragma unroll
  for (int t = 0; t < 4; ++t) {
    const int n = n0 + t * 16 + nlo;
#pragma unroll
    for (int r = 0; r < 8; ++r) {
      const size_t o = (size_t)(mb + r) * DD + n;
      y[o] = x[o] + acc[t][r];
    }
  }
}

extern "C" void kernel_launch(void* const* d_in, const int* in_sizes, int n_in,
                              void* d_out, int out_size, void* d_ws, size_t ws_size,
                              hipStream_t stream) {
  const float* x      = (const float*)d_in[0];
  const float* w_slow = (const float*)d_in[1];
  const float* ln_g   = (const float*)d_in[2];
  const float* ln_b   = (const float*)d_in[3];
  const float* w_out  = (const float*)d_in[4];
  const float* fw0    = (const float*)d_in[5];
  const float* vfw0   = (const float*)d_in[6];
  float* out = (float*)d_out;

  // Workspace carve-up (~79 MB; out1 is last and gets a small over-read pad)
  __bf16* xln   = (__bf16*)d_ws;                     // 4096*512
  __bf16* wsT   = xln   + (size_t)SS * BB * DD;      // 2624*512 (padded, [N][K])
  __bf16* out2m = wsT   + (size_t)NPAD * DD;         // 4096*512
  __bf16* woT   = out2m + (size_t)SS * BB * DD;      // 512*512  ([N][K])
  float*  qkvb  = (float*)(woT + (size_t)DD * DD);   // 4096*2576
  float*  out1  = qkvb + (size_t)SS * BB * NPROJ;    // 32*1024*193 (+64B DMA pad)

  float* y_out   = out;
  float* fw_out  = out + (size_t)SS * BB * DD;                        // 2,097,152
  float* vfw_out = fw_out + (size_t)BB * HH * DHH * DVV;              // +395,264

  ln_kernel      <<<(SS * BB) / 8, 256, 0, stream>>>(x, ln_g, ln_b, xln);
  tp_wslow_kernel<<<(NPAD * DD) / 256, 256, 0, stream>>>(w_slow, wsT);
  tp_wout_kernel <<<(DD * DD) / 256, 256, 0, stream>>>(w_out, woT);
  gemm_qkvb_kernel<<<(256 * 41) / 8, 256, 0, stream>>>(xln, wsT, qkvb);
  act1_kernel    <<<(SS * BB * HH) / 8, 256, 0, stream>>>(qkvb);
  scan1_kernel   <<<BB * HH, 256, 0, stream>>>(qkvb, fw0, out1, fw_out);
  act2_kernel    <<<(BB * HH * SS) / 8, 256, 0, stream>>>(out1);
  scan2_kernel   <<<BB * HH, 256, 0, stream>>>(out1, vfw0, out2m, vfw_out);
  gemm_out_kernel<<<(256 * 8) / 8, 256, 0, stream>>>(out2m, woT, x, y_out);
}
